// SparseMoEMLP_8881992368481
// MI455X (gfx1250) — matrix operand order repaired and checked by reading, not compiled, not run on hardware
//
#include <hip/hip_runtime.h>

// ---------------- problem constants (from reference) ----------------
#define TOK_N   8192            // B*S = 4*2048
#define DIM     1024            // D
#define NEXP    8               // E
#define HID     4096            // 4*D  (GLU half width)
#define UPN     8192            // 8*D  (up-proj output width)
#define BM      16              // token rows per workgroup tile
#define NTILES  (TOK_N / BM)    // 512 worst-case tiles per expert
#define XPITCH  (DIM + 8)       // 1032 bf16 elems (16B-aligned rows)
#define APITCH  (HID + 8)       // 4104 bf16 elems
#define BK      64              // staged K rows per buffer (2 WMMA K-steps)
#define WPITCH  (BK + 8)        // 72 bf16 elems per staged weight column
#define NCHUNK  128             // 8 waves * 16 output cols per chunk
#define WBUFSZ  (NCHUNK * WPITCH)   // one staging tile, bf16 elems (9216)

typedef __attribute__((ext_vector_type(16))) __bf16 v16bf;
typedef __attribute__((ext_vector_type(2)))  __bf16 v2bf;
typedef __attribute__((ext_vector_type(8)))  float  v8f;

union V16U { v16bf v; uint4 u[2]; };

// pack two fp32 into one dword of packed bf16 -> single v_cvt_pk_bf16_f32
#if __has_builtin(__builtin_amdgcn_cvt_pk_bf16_f32)
__device__ __forceinline__ unsigned int pk2(float lo, float hi) {
  auto r = __builtin_amdgcn_cvt_pk_bf16_f32(lo, hi);
  union { decltype(r) v; unsigned int u; } c;
  c.v = r;
  return c.u;
}
#else
__device__ __forceinline__ unsigned int pk2(float lo, float hi) {
  v2bf v;
  v.x = (__bf16)lo;
  v.y = (__bf16)hi;
  union { v2bf v; unsigned int u; } c;
  c.v = v;
  return c.u;
}
#endif

// A fragment (16x32 bf16, MxK): lanes 0-15 -> M=lane, K kb..kb+7 & kb+16..kb+23
// with kb = 0; lanes 16-31 -> same M, kb = 8.  (ISA 7.12.2 16-bit A layout)
__device__ __forceinline__ v16bf load_frag_a(const __bf16* row, int kbase, int lhalf) {
  V16U t;
  const __bf16* p = row + kbase + lhalf * 8;
  t.u[0] = *(const uint4*)(p);         // K kb .. kb+7
  t.u[1] = *(const uint4*)(p + 16);    // K kb+16 .. kb+23
  return t.v;
}

// B fragment (32x16 bf16, KxN) from LDS stored transposed [n][k]:
// lanes 0-15 -> N=lane, K ks..ks+7 & ks+16..ks+23 ; lanes 16-31 -> N=lane-16, K ks+8..ks+15 & ks+24..ks+31.
__device__ __forceinline__ v16bf load_frag_b(const __bf16* col, int ks, int lhalf) {
  V16U t;
  const __bf16* p = col + ks + lhalf * 8;
  t.u[0] = *(const uint4*)(p);
  t.u[1] = *(const uint4*)(p + 16);
  return t.v;
}

__device__ __forceinline__ v8f wmma_bf16f32(v16bf a, v16bf b, v8f c) {
  // (neg_a, A, neg_b, B, c_mod, C, reuse_a, reuse_b)
  return __builtin_amdgcn_wmma_f32_16x16x32_bf16(false, a, false, b, (short)0, c,
                                                 false, false);
}

// branch-free tanh-form GELU: g * sigmoid(1.5957691*(g + 0.044715 g^3))
// -> v_exp_f32 + v_rcp_f32 (TRANS ops co-execute with WMMA), no exec-mask forks
__device__ __forceinline__ float gelu_fast(float g) {
  float t = 1.5957691216057308f * (g + 0.044715f * g * g * g);
  return g * __builtin_amdgcn_rcpf(1.0f + __expf(-t));
}

// 8x(K) x 4x(N) fp32 weight patch staged per thread, stored transposed as bf16
struct WStage { float4 q[8]; };

__device__ __forceinline__ WStage load_w8(const float* __restrict__ g, int rowStride) {
  WStage s;
#pragma unroll
  for (int r = 0; r < 8; ++r)
    s.q[r] = *(const float4*)(g + (size_t)r * rowStride);
  return s;
}

__device__ __forceinline__ void store_w8(const WStage& s, __bf16* __restrict__ dst,
                                         int kr, int cb) {
#pragma unroll
  for (int u = 0; u < 4; ++u) {            // one ds_store_b128 per column
    uint4 p;
    p.x = pk2(((const float*)&s.q[0])[u], ((const float*)&s.q[1])[u]);  // K kr..kr+1
    p.y = pk2(((const float*)&s.q[2])[u], ((const float*)&s.q[3])[u]);  // K kr+2..3
    p.z = pk2(((const float*)&s.q[4])[u], ((const float*)&s.q[5])[u]);  // K kr+4..5
    p.w = pk2(((const float*)&s.q[6])[u], ((const float*)&s.q[7])[u]);  // K kr+6..7
    *(uint4*)(dst + (size_t)(cb + u) * WPITCH + kr) = p;
  }
}

// ---------------- kernel 0: zero out + counters ----------------
__global__ void moe_zero_kernel(float* __restrict__ out, int n4, int* __restrict__ cnt) {
  float4 z = make_float4(0.f, 0.f, 0.f, 0.f);
  for (int i = blockIdx.x * blockDim.x + threadIdx.x; i < n4;
       i += gridDim.x * blockDim.x)
    ((float4*)out)[i] = z;
  if (blockIdx.x == 0 && threadIdx.x < NEXP) cnt[threadIdx.x] = 0;
}

// ---------------- kernel 1: router (one wave32 per token) ----------------
__global__ __launch_bounds__(256)
void moe_router_kernel(const float* __restrict__ x, const float* __restrict__ rw,
                       const float* __restrict__ rb, int* __restrict__ cnt,
                       int* __restrict__ toks, float* __restrict__ wgts) {
  const int wave = threadIdx.x >> 5;
  const int lane = threadIdx.x & 31;
  const int t = blockIdx.x * 8 + wave;
  if (t >= TOK_N) return;

  float acc[NEXP];
#pragma unroll
  for (int e = 0; e < NEXP; ++e) acc[e] = 0.f;

  const float* xr = x + (size_t)t * DIM;
  for (int d = lane; d < DIM; d += 32) {
    float xv = xr[d];
    const float4* w4 = (const float4*)(rw + (size_t)d * NEXP);
    float4 w0 = w4[0], w1 = w4[1];
    acc[0] += xv * w0.x; acc[1] += xv * w0.y; acc[2] += xv * w0.z; acc[3] += xv * w0.w;
    acc[4] += xv * w1.x; acc[5] += xv * w1.y; acc[6] += xv * w1.z; acc[7] += xv * w1.w;
  }
#pragma unroll
  for (int off = 16; off > 0; off >>= 1)
#pragma unroll
    for (int e = 0; e < NEXP; ++e)
      acc[e] += __shfl_down(acc[e], off, 32);

  if (lane == 0) {
#pragma unroll
    for (int e = 0; e < NEXP; ++e) acc[e] += rb[e];
    int i0 = 0; float l0 = acc[0];
#pragma unroll
    for (int e = 1; e < NEXP; ++e)
      if (acc[e] > l0) { i0 = e; l0 = acc[e]; }
    int i1 = -1; float l1 = -3.4e38f;
#pragma unroll
    for (int e = 0; e < NEXP; ++e)
      if (e != i0 && acc[e] > l1) { i1 = e; l1 = acc[e]; }
    float z  = __expf(l1 - l0);          // softmax over {l0,l1}, l0 = max
    float p1 = z / (1.f + z);
    float p0 = 1.f - p1;
    int p = atomicAdd(&cnt[i0], 1);
    toks[i0 * TOK_N + p] = t;  wgts[i0 * TOK_N + p] = p0;
    p = atomicAdd(&cnt[i1], 1);
    toks[i1 * TOK_N + p] = t;  wgts[i1 * TOK_N + p] = p1;
  }
}

// ---------------- kernel 2: fused expert up->GLU->down ----------------
__global__ __launch_bounds__(256)
void moe_expert_kernel(const float* __restrict__ x,
                       const float* __restrict__ up_w, const float* __restrict__ up_b,
                       const float* __restrict__ dn_w, const float* __restrict__ dn_b,
                       const int* __restrict__ cnt, const int* __restrict__ toks,
                       const float* __restrict__ wgts, float* __restrict__ out) {
  const int e    = blockIdx.x / NTILES;
  const int tile = blockIdx.x % NTILES;
  const int nTok = cnt[e];
  const int base = tile * BM;
  if (base >= nTok) return;

  extern __shared__ __align__(16) char smem[];
  __bf16* lds_x   = (__bf16*)smem;                    // [BM][XPITCH]   33,024 B
  __bf16* lds_act = lds_x  + BM * XPITCH;             // [BM][APITCH]  131,328 B
  __bf16* lds_w   = lds_act + BM * APITCH;            // [2 buf][2 half][WBUFSZ] 73,728 B
  int*    s_tok   = (int*)(lds_w + 4 * WBUFSZ);       // [BM]
  float*  s_w     = (float*)(s_tok + BM);             // [BM]
#define WB(buf, half) (lds_w + ((buf) * 2 + (half)) * WBUFSZ)

  const int tid = threadIdx.x;
  if (tid < BM) {
    int r = base + tid;
    s_tok[tid] = (r < nTok) ? toks[e * TOK_N + r] : 0;
    s_w[tid]   = (r < nTok) ? wgts[e * TOK_N + r] : 0.f;
  }
  __syncthreads();

  // gather 16 token rows of x, fp32 -> packed bf16, one ds_store_b64 per float4
  for (int i = tid; i < BM * (DIM / 4); i += 256) {
    int row = i >> 8;              // / (DIM/4)
    int c4  = i & (DIM / 4 - 1);
    float4 v = *(const float4*)(x + (size_t)s_tok[row] * DIM + c4 * 4);
    uint2 p; p.x = pk2(v.x, v.y); p.y = pk2(v.z, v.w);
    *(uint2*)(lds_x + row * XPITCH + c4 * 4) = p;
  }

  const int wave  = tid >> 5;
  const int lane  = tid & 31;
  const int lhalf = lane >> 4;     // 0 or 1
  const int lm    = lane & 15;
  const int kr    = 8 * wave;      // staged K rows kr..kr+7
  const int cb    = lane * 4;      // staged N cols cb..cb+3
  const size_t upWbase = (size_t)e * DIM * UPN;
  const size_t dnWbase = (size_t)e * HID * DIM;
  const __bf16* xrow = lds_x + lm * XPITCH;
  const int bcol = (wave * 16 + lm) * WPITCH;

  // ---------------- UP proj + GLU into lds_act ----------------
  for (int chunk = 0; chunk < HID / NCHUNK; ++chunk) {   // 32 chunks of 128 cols
    const int na = chunk * NCHUNK;       // 'a' half columns
    const int ng = na + HID;             // 'g' half columns
    v8f acc_a = {}, acc_g = {};
    const float* gA0 = up_w + upWbase + (size_t)kr * UPN + na + cb;
    const float* gG0 = up_w + upWbase + (size_t)kr * UPN + ng + cb;

    __syncthreads();                     // buf0 free of prior readers
    {                                    // prologue: stage k0 = 0 into buf 0
      WStage sa = load_w8(gA0, UPN);
      WStage sg = load_w8(gG0, UPN);
      store_w8(sa, WB(0, 0), kr, cb);
      store_w8(sg, WB(0, 1), kr, cb);
    }
    int cur = 0;
    for (int k0 = 0; k0 < DIM; k0 += BK) {       // 16 staged steps
      __syncthreads();                   // staged tile 'cur' visible to all
      WStage sa, sg;
      const bool more = (k0 + BK) < DIM;
      if (more) {                        // issue next-step global loads early
        const float* gA = gA0 + (size_t)(k0 + BK) * UPN;
        const float* gG = gG0 + (size_t)(k0 + BK) * UPN;
        __builtin_prefetch(gA + (size_t)BK * UPN, 0, 1);
        __builtin_prefetch(gG + (size_t)BK * UPN, 0, 1);
        sa = load_w8(gA, UPN);
        sg = load_w8(gG, UPN);
      }
#pragma unroll
      for (int ks = 0; ks < BK; ks += 32) {      // 2 WMMA K-steps per stage
        v16bf af = load_frag_a(xrow, k0 + ks, lhalf);
        v16bf ba = load_frag_b(WB(cur, 0) + bcol, ks, lhalf);
        v16bf bg = load_frag_b(WB(cur, 1) + bcol, ks, lhalf);
        acc_a = wmma_bf16f32(af, ba, acc_a);
        acc_g = wmma_bf16f32(af, bg, acc_g);
      }
      if (more) {                        // convert+store into the other buffer
        store_w8(sa, WB(cur ^ 1, 0), kr, cb);
        store_w8(sg, WB(cur ^ 1, 1), kr, cb);
      }
      cur ^= 1;
    }

    // bias + GLU, store bf16 activation tile
    const int col = wave * 16 + lm;
    const float biasA = up_b[(size_t)e * UPN + na + col];
    const float biasG = up_b[(size_t)e * UPN + ng + col];
#pragma unroll
    for (int r = 0; r < 8; ++r) {
      int m = r + 8 * lhalf;
      float a = acc_a[r] + biasA;
      float g = acc_g[r] + biasG;
      lds_act[m * APITCH + na + col] = (__bf16)(a * gelu_fast(g));
    }
  }
  __syncthreads();                       // all activations visible

  // ---------------- DOWN proj + weighted combine ----------------
  for (int chunk = 0; chunk < DIM / NCHUNK; ++chunk) {   // 8 chunks of 128 cols
    const int n2 = chunk * NCHUNK;
    v8f acc = {};
    const float* gD0 = dn_w + dnWbase + (size_t)kr * DIM + n2 + cb;

    __syncthreads();
    {                                    // prologue: stage k0 = 0 into buf 0
      WStage sd = load_w8(gD0, DIM);
      store_w8(sd, WB(0, 0), kr, cb);
    }
    int cur = 0;
    for (int k0 = 0; k0 < HID; k0 += BK) {       // 64 staged steps
      __syncthreads();
      WStage sd;
      const bool more = (k0 + BK) < HID;
      if (more) {
        const float* gD = gD0 + (size_t)(k0 + BK) * DIM;
        __builtin_prefetch(gD + (size_t)BK * DIM, 0, 1);
        sd = load_w8(gD, DIM);
      }
#pragma unroll
      for (int ks = 0; ks < BK; ks += 32) {
        v16bf af = load_frag_a(lds_act + lm * APITCH, k0 + ks, lhalf);
        v16bf bf = load_frag_b(WB(cur, 0) + bcol, ks, lhalf);
        acc = wmma_bf16f32(af, bf, acc);
      }
      if (more) store_w8(sd, WB(cur ^ 1, 0), kr, cb);
      cur ^= 1;
    }

    const int col = n2 + wave * 16 + lm;
    const float biasD = dn_b[(size_t)e * DIM + col];
#pragma unroll
    for (int r = 0; r < 8; ++r) {
      int m = r + 8 * lhalf;
      float y = (acc[r] + biasD) * s_w[m];
      unsafeAtomicAdd(out + (size_t)s_tok[m] * DIM + col, y);
    }
  }
#undef WB
}

// ---------------- host launch ----------------
extern "C" void kernel_launch(void* const* d_in, const int* in_sizes, int n_in,
                              void* d_out, int out_size, void* d_ws, size_t ws_size,
                              hipStream_t stream) {
  const float* x    = (const float*)d_in[0];
  const float* rw   = (const float*)d_in[1];
  const float* rb   = (const float*)d_in[2];
  const float* up_w = (const float*)d_in[3];
  const float* up_b = (const float*)d_in[4];
  const float* dn_w = (const float*)d_in[5];
  const float* dn_b = (const float*)d_in[6];
  float* out = (float*)d_out;

  int*   cnt  = (int*)d_ws;
  int*   toks = (int*)((char*)d_ws + 256);
  float* wgts = (float*)((char*)d_ws + 256 + (size_t)NEXP * TOK_N * sizeof(int));

  // x-tile + act-tile + 2x double-buffered 64-deep staging tiles + token ids/weights
  const size_t smemBytes =
      (size_t)2 * (BM * XPITCH + BM * APITCH + 4 * WBUFSZ) +
      BM * sizeof(int) + BM * sizeof(float);   // 238,208 B of the 320 KB WGP LDS

  moe_zero_kernel<<<1024, 256, 0, stream>>>(out, TOK_N * DIM / 4, cnt);
  moe_router_kernel<<<TOK_N / 8, 256, 0, stream>>>(x, rw, rb, cnt, toks, wgts);
  moe_expert_kernel<<<NEXP * NTILES, 256, smemBytes, stream>>>(
      x, up_w, up_b, dn_w, dn_b, cnt, toks, wgts, out);
}
